// MahalanobisMetricLoss_87909390615170
// MI455X (gfx1250) — compile-verified, matrix operand order checked by reading, not run on hardware
//
#include <hip/hip_runtime.h>
#include <hip/hip_bf16.h>
#include <math.h>

typedef _Float16 v8h  __attribute__((ext_vector_type(8)));
typedef _Float16 v16h __attribute__((ext_vector_type(16)));
typedef float    v8f  __attribute__((ext_vector_type(8)));

#define BN     4096
#define DIM    512
#define PADK   40          // 32 halves + 8 pad -> 80B row stride, keeps 16B alignment
#define LO_M   0.56f       // MARGIN - EXTRA_MARGIN
#define HI_M   0.64f       // MARGIN + EXTRA_MARGIN
#define MARG   0.60f

// ---------------------------------------------------------------------------
// CDNA5 async global->LDS staging (ASYNCcnt-tracked), guarded so the file
// still compiles on toolchains without the builtin (falls back to ds_store).
// Builtin signature (from hipcc diagnostic): (int4 AS1*, int4 AS3*, Imm, Imm).
// ---------------------------------------------------------------------------
#if defined(__has_builtin)
#if __has_builtin(__builtin_amdgcn_global_load_async_to_lds_b128)
#define USE_ASYNC_LDS 1
#endif
#endif

#ifdef USE_ASYNC_LDS
typedef int v4i_t __attribute__((vector_size(16)));
typedef __attribute__((address_space(1))) v4i_t glob_v4i;
typedef __attribute__((address_space(3))) v4i_t lds_v4i;
// generic LDS pointer keeps the LDS byte offset in its low 32 bits (ISA 10.2);
// generic global pointer value == AS(1) address. Integer round-trip casts.
__device__ __forceinline__ void stage16(const _Float16* gsrc, _Float16* ldst) {
  glob_v4i* g = (glob_v4i*)(unsigned long long)(uintptr_t)gsrc;
  lds_v4i*  l = (lds_v4i*)(unsigned)(uintptr_t)ldst;
  __builtin_amdgcn_global_load_async_to_lds_b128(g,     l,     0, 0);
  __builtin_amdgcn_global_load_async_to_lds_b128(g + 1, l + 1, 0, 0);
}
__device__ __forceinline__ void wait_stage() {
#if __has_builtin(__builtin_amdgcn_s_wait_asynccnt)
  __builtin_amdgcn_s_wait_asynccnt(0);
#else
  asm volatile("s_wait_asynccnt 0x0" ::: "memory");
#endif
}
#else
__device__ __forceinline__ void stage16(const _Float16* gsrc, _Float16* ldst) {
  *(uint4*)(ldst)     = *(const uint4*)(gsrc);
  *(uint4*)(ldst + 8) = *(const uint4*)(gsrc + 8);
}
__device__ __forceinline__ void wait_stage() {}  // ds_store waits handled at barrier
#endif

// ---------------------------------------------------------------------------
// Kernel 1: fp32 -> fp16 conversion + row squared-norms computed from the
// CONVERTED values (keeps the WMMA diagonal consistent: sq[i][i] ~ 0).
// ---------------------------------------------------------------------------
__global__ __launch_bounds__(256) void prep_kernel(const float* __restrict__ X,
                                                   _Float16* __restrict__ Xh,
                                                   float* __restrict__ sn) {
  const int lane = threadIdx.x & 31;
  const int wid  = threadIdx.x >> 5;
  const int row  = blockIdx.x * 8 + wid;
  const float*  xr = X  + (size_t)row * DIM;
  _Float16*     hr = Xh + (size_t)row * DIM;
  float s = 0.f;
  for (int c = lane; c < DIM; c += 32) {
    _Float16 h = (_Float16)xr[c];
    hr[c] = h;
    float hv = (float)h;
    s += hv * hv;
  }
  for (int off = 16; off > 0; off >>= 1) s += __shfl_down(s, off, 32);
  if (lane == 0) sn[row] = s;
}

// ---------------------------------------------------------------------------
// Kernel 2: tiled X·Xᵀ via v_wmma_f32_16x16x32_f16 with fused distance/mask
// epilogue. 128x128 tile per WG, 8 waves (4M x 2N), each wave 32x64 = 2x4
// accumulators. Double-buffered LDS: async global->LDS fetch of K-step k+1
// overlaps the WMMA work of K-step k; one barrier per K-step.
// ---------------------------------------------------------------------------
__global__ __launch_bounds__(256) void
tile_kernel(const _Float16* __restrict__ Xh, const float* __restrict__ sn,
            const int* __restrict__ tgt, float* __restrict__ posPart,
            unsigned* __restrict__ nRight, unsigned* __restrict__ cnt,
            float* __restrict__ list, unsigned cap) {
  __shared__ __align__(16) _Float16 As[2][128 * PADK];
  __shared__ __align__(16) _Float16 Bs[2][128 * PADK];
  __shared__ float snI[128], snJ[128];
  __shared__ int   tIs[128], tJs[128];
  __shared__ float red[256];
  __shared__ unsigned redR;

  const int tid   = threadIdx.x;
  const int iBase = blockIdx.y << 7;
  const int jBase = blockIdx.x << 7;

  if (tid < 128) { snI[tid] = sn[iBase + tid]; tIs[tid] = tgt[iBase + tid]; }
  else { const int u = tid - 128; snJ[u] = sn[jBase + u]; tJs[u] = tgt[jBase + u]; }
  if (tid == 0) redR = 0;

  const int lane  = tid & 31;
  const int wid   = tid >> 5;
  const int waveM = wid >> 1;        // 0..3  (rows, 32 each)
  const int waveN = wid & 1;         // 0..1  (cols, 64 each)
  const int lrow  = lane & 15;
  const int lhi   = lane >> 4;
  const int koff  = lhi << 3;        // K offset in halves per ISA 16-bit A/B layout

  v8f acc[2][4] = {};

  // cooperative staging: thread t copies 16 halves (32B) per operand per step
  const int gRow = tid >> 1;               // 0..127
  const int gSeg = (tid & 1) << 4;         // 0 or 16 halves
  const _Float16* aSrc = Xh + (size_t)(iBase + gRow) * DIM + gSeg;
  const _Float16* bSrc = Xh + (size_t)(jBase + gRow) * DIM + gSeg;
  const int dOff = gRow * PADK + gSeg;

  auto compute = [&](const _Float16* Ab, const _Float16* Bb) {
    v16h a[2], b[4];
#pragma unroll
    for (int fm = 0; fm < 2; ++fm) {
      const _Float16* p = Ab + (waveM * 32 + fm * 16 + lrow) * PADK + koff;
      v8h x0 = *(const v8h*)(p);
      v8h x1 = *(const v8h*)(p + 16);
      a[fm] = __builtin_shufflevector(x0, x1, 0,1,2,3,4,5,6,7,8,9,10,11,12,13,14,15);
    }
#pragma unroll
    for (int fn = 0; fn < 4; ++fn) {
      const _Float16* p = Bb + (waveN * 64 + fn * 16 + lrow) * PADK + koff;
      v8h x0 = *(const v8h*)(p);
      v8h x1 = *(const v8h*)(p + 16);
      b[fn] = __builtin_shufflevector(x0, x1, 0,1,2,3,4,5,6,7,8,9,10,11,12,13,14,15);
    }
#pragma unroll
    for (int fm = 0; fm < 2; ++fm)
#pragma unroll
      for (int fn = 0; fn < 4; ++fn)
        acc[fm][fn] = __builtin_amdgcn_wmma_f32_16x16x32_f16(
            false, a[fm], false, b[fn], (short)0, acc[fm][fn], false, false);
  };

  // prologue: stage K-step 0 into buffer 0
  stage16(aSrc, &As[0][dOff]);
  stage16(bSrc, &Bs[0][dOff]);
  wait_stage();
  __syncthreads();

  for (int k0 = 0; k0 < DIM; k0 += 64) {
    // stage k0+32 into buf1 while computing on buf0
    if (k0 + 32 < DIM) {
      stage16(aSrc + k0 + 32, &As[1][dOff]);
      stage16(bSrc + k0 + 32, &Bs[1][dOff]);
    }
    compute(As[0], Bs[0]);
    wait_stage();
    __syncthreads();

    // stage k0+64 into buf0 while computing on buf1
    if (k0 + 64 < DIM) {
      stage16(aSrc + k0 + 64, &As[0][dOff]);
      stage16(bSrc + k0 + 64, &Bs[0][dOff]);
    }
    compute(As[1], Bs[1]);
    wait_stage();
    __syncthreads();
  }

  // fused epilogue: G -> distance -> masked reductions (no dist matrix in HBM)
  float    lPos   = 0.f;
  unsigned lRight = 0;
#pragma unroll
  for (int fm = 0; fm < 2; ++fm) {
#pragma unroll
    for (int fn = 0; fn < 4; ++fn) {
      const int   jLoc = waveN * 64 + fn * 16 + lrow;  // C layout: N = lane%16
      const float snj  = snJ[jLoc];
      const int   tj   = tJs[jLoc];
#pragma unroll
      for (int v = 0; v < 8; ++v) {
        const int iLoc = waveM * 32 + fm * 16 + lhi * 8 + v;  // M = vgpr + 8*(lane>=16)
        const float g  = acc[fm][fn][v];
        float sq = snI[iLoc] + snj - 2.0f * g;
        const bool pos = (tIs[iLoc] == tj);
        const float d  = (sq > 0.f) ? __builtin_amdgcn_sqrtf(sq) : 0.f;
        if (pos) {
          if (d > LO_M) lPos += d - LO_M;
          if (d < MARG) lRight++;
        } else {
          if (d >= MARG) lRight++;
          if (d < HI_M) {  // candidate for neg_loss; thr resolved in finalize
            const unsigned idx = atomicAdd(cnt, 1u);
            if (idx < cap) list[idx] = d;
          }
        }
      }
    }
  }

  red[tid] = lPos;
  atomicAdd(&redR, lRight);  // integer: exact, order-independent
  __syncthreads();
  for (int s = 128; s > 0; s >>= 1) {
    if (tid < s) red[tid] += red[tid + s];
    __syncthreads();
  }
  if (tid == 0) {
    posPart[blockIdx.y * gridDim.x + blockIdx.x] = red[0];
    atomicAdd(nRight, redR);
  }
}

// ---------------------------------------------------------------------------
// Kernel 3 (1 block): class histogram -> num_pairs; deterministic sum of the
// 1024 block partials; bit-exact k-th order-statistic selection over the
// candidate list (only if > num_pairs candidates exist below HI_M); outputs.
// ---------------------------------------------------------------------------
__global__ __launch_bounds__(256) void
finalize_kernel(const int* __restrict__ tgt, const float* __restrict__ posPart,
                const unsigned* __restrict__ nRight, const unsigned* __restrict__ cnt,
                const float* __restrict__ list, unsigned cap,
                float* __restrict__ out) {
  __shared__ unsigned bins[64];
  __shared__ float    red[256];
  __shared__ unsigned uCnt;
  __shared__ unsigned sNumPairs;
  __shared__ float    sPosLoss;

  const int tid = threadIdx.x;
  if (tid < 64) bins[tid] = 0;
  __syncthreads();
  for (int t = tid; t < BN; t += 256) atomicAdd(&bins[tgt[t]], 1u);
  __syncthreads();
  if (tid == 0) {
    unsigned long long sp = 0;
    for (int c = 0; c < 64; ++c)
      sp += (unsigned long long)bins[c] * (unsigned long long)bins[c];
    sNumPairs = (unsigned)((sp - (unsigned long long)BN) >> 1);
  }
  __syncthreads();
  const unsigned numPairs = sNumPairs;
  unsigned C = *cnt;
  if (C > cap) C = cap;

  // deterministic sum of pos_loss partials
  float lp = 0.f;
  for (int b = tid; b < 1024; b += 256) lp += posPart[b];
  red[tid] = lp;
  __syncthreads();
  for (int s = 128; s > 0; s >>= 1) {
    if (tid < s) red[tid] += red[tid + s];
    __syncthreads();
  }
  if (tid == 0) sPosLoss = red[0];
  __syncthreads();

  // thr = sorted(neg dists)[num_pairs]; only binds if > num_pairs entries < HI_M
  unsigned thrBits = 0xFFFFFFFFu;  // "no constraint": d < HI_M already enforced
  if (C > numPairs) {
    unsigned lo = 0, hiB = __float_as_uint(HI_M);
    while (lo < hiB) {
      const unsigned mid = lo + ((hiB - lo) >> 1);
      if (tid == 0) uCnt = 0;
      __syncthreads();
      unsigned c = 0;
      for (unsigned idx = tid; idx < C; idx += 256)
        c += (__float_as_uint(list[idx]) <= mid) ? 1u : 0u;
      atomicAdd(&uCnt, c);
      __syncthreads();
      const unsigned total = uCnt;
      __syncthreads();
      if (total >= numPairs + 1u) hiB = mid; else lo = mid + 1u;
    }
    thrBits = lo;  // exact bit pattern of thr; preserves strict d < thr
  }

  float ln = 0.f;
  for (unsigned idx = tid; idx < C; idx += 256) {
    const float dv = list[idx];
    if (__float_as_uint(dv) < thrBits) ln += HI_M - dv;
  }
  red[tid] = ln;
  __syncthreads();
  for (int s = 128; s > 0; s >>= 1) {
    if (tid < s) red[tid] += red[tid + s];
    __syncthreads();
  }
  if (tid == 0) {
    const float negLoss = red[0];
    out[0] = (sPosLoss + negLoss) / (2.0f * (float)numPairs);
    out[1] = (float)(*nRight) / ((float)BN * (float)BN);
  }
}

// ---------------------------------------------------------------------------
// Workspace layout:
//   [0,4)        nRight (u32)        [4,8)  list count (u32)   (zeroed/launch)
//   [64,4160)    posPart[1024] f32   (fully overwritten each launch)
//   [4160,20544) sq_norms[4096] f32
//   [20544, +4MB) Xh fp16 (4096x512)
//   [rest)       neg-candidate list (f32)
// ---------------------------------------------------------------------------
extern "C" void kernel_launch(void* const* d_in, const int* in_sizes, int n_in,
                              void* d_out, int out_size, void* d_ws, size_t ws_size,
                              hipStream_t stream) {
  const float* X   = (const float*)d_in[0];
  const int*   tgt = (const int*)d_in[1];
  float*       out = (float*)d_out;

  char* ws = (char*)d_ws;
  unsigned* nRight  = (unsigned*)(ws + 0);
  unsigned* cnt     = (unsigned*)(ws + 4);
  float*    posPart = (float*)(ws + 64);
  float*    sn      = (float*)(ws + 64 + 1024 * 4);            // 4160
  _Float16* Xh      = (_Float16*)(ws + 4160 + (size_t)BN * 4); // 20544
  const size_t listOff = 20544 + (size_t)BN * DIM * 2;         // 4214848
  float*    list    = (float*)(ws + listOff);
  const unsigned cap =
      (ws_size > listOff + 4) ? (unsigned)((ws_size - listOff) / 4) : 0u;

  (void)hipMemsetAsync(ws, 0, 64, stream);  // zero counters (graph-capture safe)
  prep_kernel<<<BN / 8, 256, 0, stream>>>(X, Xh, sn);
  tile_kernel<<<dim3(BN / 128, BN / 128), 256, 0, stream>>>(
      Xh, sn, tgt, posPart, nRight, cnt, list, cap);
  finalize_kernel<<<1, 256, 0, stream>>>(tgt, posPart, nRight, cnt, list, cap, out);
}
